// EncoderRNN_32160715113133
// MI455X (gfx1250) — matrix operand order repaired and checked by reading, not compiled, not run on hardware
//
#include <hip/hip_runtime.h>
#include <hip/hip_bf16.h>
#include <math.h>

// GRU: T=256, B=128, I=H=1024. fp32 in/out, f16 WMMA with f32 accumulation.
// CDNA5 paths: v_wmma_f32_16x16x32_f16 + global_load_async_to_lds_b128
// double-buffered staging (ASYNCcnt pipelining).

typedef __attribute__((ext_vector_type(16))) _Float16 v16h;
typedef __attribute__((ext_vector_type(8)))  _Float16 v8h;
typedef __attribute__((ext_vector_type(8)))  float    v8f;

#define Tn 256
#define Bn 128
#define In 1024
#define Hn 1024
#define G3 3072
#define TB (Tn * Bn)   // 32768
#define LDA 40         // LDS row stride in halves (padded: 32 + 8)

// 16B global -> LDS async copy (per active lane), tracked by ASYNCcnt.
__device__ __forceinline__ void async_cp_b128(const void* gptr, void* lptr) {
  unsigned lds = (unsigned)(size_t)lptr;                 // LDS byte offset
  unsigned long long ga = (unsigned long long)(size_t)gptr;
  asm volatile("global_load_async_to_lds_b128 %0, %1, off"
               :: "v"(lds), "v"(ga) : "memory");
}

// ---------------------------------------------------------------- converts
__global__ void cvt_f32_to_f16(const float* __restrict__ src,
                               _Float16* __restrict__ dst, int n) {
  int i = blockIdx.x * blockDim.x + threadIdx.x;
  if (i < n) dst[i] = (_Float16)src[i];
}

__global__ void zero_h(_Float16* __restrict__ hA, float* __restrict__ hF) {
  int i = blockIdx.x * blockDim.x + threadIdx.x;   // Bn*Hn threads
  hA[i] = (_Float16)0.0f;
  hF[i] = 0.0f;
}

// ------------------------------------------------- input projection GEMM
// xg[M=32768, N=3072] = Xf16[M,1024] * Wih[N,1024]^T + b_ih
// Block: 128M x 64N, 8 waves; wave w -> M tile w, 4 N tiles.
__global__ __launch_bounds__(256) void gemm_xg(
    const _Float16* __restrict__ X,    // [32768,1024]
    const _Float16* __restrict__ W,    // [3072,1024]
    const float*    __restrict__ bias, // [3072]
    float*          __restrict__ xg)   // [32768,3072]
{
  __shared__ _Float16 sA[2][128 * LDA];   // 2 x 10240 B
  __shared__ _Float16 sB[2][64 * LDA];    // 2 x  5120 B
  const int tid  = threadIdx.x;
  const int lane = tid & 31;
  const int w    = tid >> 5;
  const int m0   = blockIdx.y * 128;
  const int n0   = blockIdx.x * 64;

  // Stage one 32-wide K slice into LDS buffer `buf` (3 async copies/thread).
  auto stage = [&](int buf, int kc) {
    int chunk = tid;
    #pragma unroll
    for (int r = 0; r < 2; ++r, chunk += 256) {   // A: 512 x 16B chunks
      int row = chunk >> 2;
      int c8  = (chunk & 3) * 8;
      async_cp_b128(X + (size_t)(m0 + row) * In + kc + c8,
                    &sA[buf][row * LDA + c8]);
    }
    {                                             // B: 256 x 16B chunks
      int row = tid >> 2;
      int c8  = (tid & 3) * 8;
      async_cp_b128(W + (size_t)(n0 + row) * In + kc + c8,
                    &sB[buf][row * LDA + c8]);
    }
  };

  v8f acc[4] = {};
  const int NK = In / 32;
  stage(0, 0);

  for (int i = 0; i < NK; ++i) {
    int cur = i & 1;
    if (i + 1 < NK) {
      stage(cur ^ 1, (i + 1) * 32);   // prefetch next slice
      // just issued 3 -> waiting <=3 means slice i's copies have landed
      asm volatile("s_wait_asynccnt 0x3" ::: "memory");
    } else {
      asm volatile("s_wait_asynccnt 0x0" ::: "memory");
    }
    __syncthreads();   // data-ready across waves

    // A fragment (16x32 f16, ISA layout): lane L: M=L&15,
    // halves 0-7 -> K=8*(L>>4).., halves 8-15 -> K=16+8*(L>>4)..
    v16h a;
    {
      int row = 16 * w + (lane & 15);
      int off = 8 * (lane >> 4);
      v8h lo = *(const v8h*)(&sA[cur][row * LDA + off]);
      v8h hi = *(const v8h*)(&sA[cur][row * LDA + 16 + off]);
      #pragma unroll
      for (int k = 0; k < 8; ++k) { a[k] = lo[k]; a[8 + k] = hi[k]; }
    }
    #pragma unroll
    for (int nt = 0; nt < 4; ++nt) {
      // B fragment (32x16 f16): lane L: N=L&15, halves 0-7 -> K=8*(L>>4).., halves 8-15 -> K=16+8*(L>>4)..
      v16h b;
      int row = 16 * nt + (lane & 15);
      int off = 8 * (lane >> 4);
      v8h lo = *(const v8h*)(&sB[cur][row * LDA + off]);
      v8h hi = *(const v8h*)(&sB[cur][row * LDA + 16 + off]);
      #pragma unroll
      for (int k = 0; k < 8; ++k) { b[k] = lo[k]; b[8 + k] = hi[k]; }
      acc[nt] = __builtin_amdgcn_wmma_f32_16x16x32_f16(
          false, a, false, b, (short)0, acc[nt], false, false);
    }
    __syncthreads();   // read-done: safe to overwrite buffer cur^1 next iter
  }

  // Epilogue: C layout — VGPR j: lanes 0-15 M=j, lanes 16-31 M=8+j; N=lane&15
  #pragma unroll
  for (int nt = 0; nt < 4; ++nt) {
    int n = n0 + 16 * nt + (lane & 15);
    float bn = bias[n];
    #pragma unroll
    for (int j = 0; j < 8; ++j) {
      int m = m0 + 16 * w + j + 8 * (lane >> 4);
      xg[(size_t)m * G3 + n] = acc[nt][j] + bn;
    }
  }
}

// ------------------------------------------------------- recurrent step
// Block: all 128 rows x one 16-wide H slice, computed in all 3 gate chunks.
// 8 waves: wave w -> M tile w; 3 accumulators (r,z,n chunks).
__global__ __launch_bounds__(256) void gru_step(
    const _Float16* __restrict__ hA,      // [128,1024] f16 (h_t)
    const float*    __restrict__ hF,      // [128,1024] f32 (h_t)
    const _Float16* __restrict__ Wh,      // [3072,1024] f16
    const float*    __restrict__ b_hh,    // [3072]
    const float*    __restrict__ xg_t,    // [128,3072]
    float*          __restrict__ out_t,   // [128,1024]
    _Float16*       __restrict__ hA_next, // [128,1024]
    float*          __restrict__ hF_next) // [128,1024]
{
  __shared__ _Float16 sA[2][128 * LDA];  // 2 x 10240 B
  __shared__ _Float16 sB[2][48 * LDA];   // 2 x  3840 B
  const int tid  = threadIdx.x;
  const int lane = tid & 31;
  const int w    = tid >> 5;
  const int n0   = blockIdx.x * 16;

  // Waves 0-5 issue 3 async copies per slice, waves 6-7 issue 2.
  auto stage = [&](int buf, int kc) {
    int chunk = tid;
    #pragma unroll
    for (int r = 0; r < 2; ++r, chunk += 256) {   // A = h: 512 x 16B chunks
      int row = chunk >> 2;
      int c8  = (chunk & 3) * 8;
      async_cp_b128(hA + (size_t)row * Hn + kc + c8,
                    &sA[buf][row * LDA + c8]);
    }
    if (tid < 192) {                              // B: 3 chunks x 16 Wh rows
      int row  = tid >> 2;        // 0..47
      int c8   = (tid & 3) * 8;
      int ch   = row >> 4;        // gate chunk 0..2
      int rloc = row & 15;
      async_cp_b128(Wh + (size_t)(ch * Hn + n0 + rloc) * Hn + kc + c8,
                    &sB[buf][row * LDA + c8]);
    }
  };

  v8f acc[3] = {};
  const int NK = Hn / 32;
  stage(0, 0);

  for (int i = 0; i < NK; ++i) {
    int cur = i & 1;
    if (i + 1 < NK) {
      stage(cur ^ 1, (i + 1) * 32);
      if (w < 6) asm volatile("s_wait_asynccnt 0x3" ::: "memory");
      else       asm volatile("s_wait_asynccnt 0x2" ::: "memory");
    } else {
      asm volatile("s_wait_asynccnt 0x0" ::: "memory");
    }
    __syncthreads();

    v16h a;
    {
      int row = 16 * w + (lane & 15);
      int off = 8 * (lane >> 4);
      v8h lo = *(const v8h*)(&sA[cur][row * LDA + off]);
      v8h hi = *(const v8h*)(&sA[cur][row * LDA + 16 + off]);
      #pragma unroll
      for (int k = 0; k < 8; ++k) { a[k] = lo[k]; a[8 + k] = hi[k]; }
    }
    #pragma unroll
    for (int c = 0; c < 3; ++c) {
      v16h b;
      int row = 16 * c + (lane & 15);
      int off = 8 * (lane >> 4);
      v8h lo = *(const v8h*)(&sB[cur][row * LDA + off]);
      v8h hi = *(const v8h*)(&sB[cur][row * LDA + 16 + off]);
      #pragma unroll
      for (int k = 0; k < 8; ++k) { b[k] = lo[k]; b[8 + k] = hi[k]; }
      acc[c] = __builtin_amdgcn_wmma_f32_16x16x32_f16(
          false, a, false, b, (short)0, acc[c], false, false);
    }
    __syncthreads();
  }

  // Gate epilogue — same (m,n) element of all three chunks sits in the
  // same lane/VGPR slot, so this is register-local per lane.
  int col = n0 + (lane & 15);
  float bhr = b_hh[col];
  float bhz = b_hh[Hn + col];
  float bhn = b_hh[2 * Hn + col];
  #pragma unroll
  for (int j = 0; j < 8; ++j) {
    int m = 16 * w + j + 8 * (lane >> 4);
    const float* xrow = xg_t + (size_t)m * G3;
    float xr = xrow[col];
    float xz = xrow[Hn + col];
    float xn = xrow[2 * Hn + col];
    float hr = acc[0][j] + bhr;
    float hz = acc[1][j] + bhz;
    float hn = acc[2][j] + bhn;
    float r  = 1.0f / (1.0f + __expf(-(xr + hr)));
    float z  = 1.0f / (1.0f + __expf(-(xz + hz)));
    float nn = tanhf(xn + r * hn);
    float hp = hF[(size_t)m * Hn + col];
    float hv = (1.0f - z) * nn + z * hp;
    size_t o = (size_t)m * Hn + col;
    out_t[o]   = hv;
    hF_next[o] = hv;
    hA_next[o] = (_Float16)hv;
  }
}

// ---------------------------------------------------------------- launch
static inline size_t align_up(size_t x) { return (x + 255) & ~(size_t)255; }

extern "C" void kernel_launch(void* const* d_in, const int* in_sizes, int n_in,
                              void* d_out, int out_size, void* d_ws, size_t ws_size,
                              hipStream_t stream) {
  (void)in_sizes; (void)n_in; (void)out_size; (void)ws_size;
  const float* x    = (const float*)d_in[0];   // [256,128,1024]
  const float* w_ih = (const float*)d_in[1];   // [3072,1024]
  const float* w_hh = (const float*)d_in[2];   // [3072,1024]
  const float* b_ih = (const float*)d_in[3];   // [3072]
  const float* b_hh = (const float*)d_in[4];   // [3072]
  float* out = (float*)d_out;                  // [256,128,1024]

  char* ws = (char*)d_ws;
  _Float16* xh   = (_Float16*)ws; ws += align_up((size_t)TB * In * 2);
  _Float16* wihh = (_Float16*)ws; ws += align_up((size_t)G3 * In * 2);
  _Float16* whhh = (_Float16*)ws; ws += align_up((size_t)G3 * Hn * 2);
  float*    xg   = (float*)ws;    ws += align_up((size_t)TB * G3 * 4);
  _Float16* hA0  = (_Float16*)ws; ws += align_up((size_t)Bn * Hn * 2);
  _Float16* hA1  = (_Float16*)ws; ws += align_up((size_t)Bn * Hn * 2);
  float*    hF0  = (float*)ws;    ws += align_up((size_t)Bn * Hn * 4);
  float*    hF1  = (float*)ws;    ws += align_up((size_t)Bn * Hn * 4);

  // f32 -> f16 conversions
  {
    int n = TB * In;
    cvt_f32_to_f16<<<(n + 255) / 256, 256, 0, stream>>>(x, xh, n);
  }
  {
    int n = G3 * In;
    cvt_f32_to_f16<<<(n + 255) / 256, 256, 0, stream>>>(w_ih, wihh, n);
    cvt_f32_to_f16<<<(n + 255) / 256, 256, 0, stream>>>(w_hh, whhh, n);
  }
  zero_h<<<(Bn * Hn) / 256, 256, 0, stream>>>(hA0, hF0);

  // Big input-projection GEMM: grid (N tiles=48, M tiles=256)
  gemm_xg<<<dim3(G3 / 64, TB / 128), 256, 0, stream>>>(xh, wihh, b_ih, xg);

  // Sequential recurrence: 256 dependent step kernels (graph-friendly)
  _Float16* hA[2] = {hA0, hA1};
  float*    hF[2] = {hF0, hF1};
  for (int t = 0; t < Tn; ++t) {
    int cur = t & 1, nxt = cur ^ 1;
    gru_step<<<Hn / 16, 256, 0, stream>>>(
        hA[cur], hF[cur], whhh, b_hh,
        xg + (size_t)t * Bn * G3,
        out + (size_t)t * Bn * Hn,
        hA[nxt], hF[nxt]);
  }
}
